// M_47837345743451
// MI455X (gfx1250) — matrix operand order repaired and checked by reading, not compiled, not run on hardware
//
#include <hip/hip_runtime.h>

typedef _Float16 h4  __attribute__((ext_vector_type(4)));
typedef _Float16 h8  __attribute__((ext_vector_type(8)));
typedef _Float16 v16h __attribute__((ext_vector_type(16)));
typedef float    v8f __attribute__((ext_vector_type(8)));
typedef float    f4  __attribute__((ext_vector_type(4)));

#define SEQ 384
#define DIM 64
#define VT_STRIDE 392   // halfs per Vt row (384 + 8 pad)
#define P_STRIDE  400   // halfs per probs row (384 + 16 pad)
#define LOG2E 1.44269504088896340736f

__global__ __launch_bounds__(128) void attn_pv_kernel(
    const float* __restrict__ scores,   // [B*H, S, S]
    const float* __restrict__ V,        // [B*H, S, D]
    const float* __restrict__ bias,     // [S]
    float* __restrict__ out)            // [B*H, S, D]
{
    __shared__ _Float16 s_vt[DIM * VT_STRIDE];        // V transposed, f16: s_vt[d][k]
    __shared__ _Float16 s_probs[4 * 16 * P_STRIDE];   // per-wave 16 rows of probs (f16)
    __shared__ float    s_rowinv[64];                 // 1/rowsum per query row

    const int t    = threadIdx.x;
    const int lane = t & 31;
    const int w    = t >> 5;            // wave id 0..3
    const int bh   = blockIdx.x / 6;    // 0..255
    const int qblk = blockIdx.x % 6;
    const int q0   = qblk * 64;

    // ---- Stage 1: V (f32 [S][D]) -> LDS transposed f16 Vt[d][k], coalesced global reads
    {
        const float* vbase = V + (size_t)bh * SEQ * DIM;
        #pragma unroll 4
        for (int it = 0; it < 48; ++it) {             // 48*128*4 = 24576 elements
            int idx = (it * 128 + t) * 4;
            f4 vv = *(const f4*)(vbase + idx);
            int k = idx >> 6;          // row of V (= K index)
            int d = idx & 63;          // column of V (multiple of 4)
            s_vt[(d + 0) * VT_STRIDE + k] = (_Float16)vv[0];
            s_vt[(d + 1) * VT_STRIDE + k] = (_Float16)vv[1];
            s_vt[(d + 2) * VT_STRIDE + k] = (_Float16)vv[2];
            s_vt[(d + 3) * VT_STRIDE + k] = (_Float16)vv[3];
        }
    }
    __syncthreads();

    // ---- Stage 2: softmax for this wave's 16 rows; probs (f16) -> LDS in A-friendly layout
    f4 bv[3];
    #pragma unroll
    for (int i = 0; i < 3; ++i)
        bv[i] = *(const f4*)(bias + (i * 32 + lane) * 4);

    const float* srow0 = scores + ((size_t)bh * SEQ + q0 + w * 16) * SEQ;
    _Float16* prow_base = &s_probs[w * 16 * P_STRIDE];

    for (int r = 0; r < 16; ++r) {
        f4 sv[3];
        float m = -3.402823466e38f;
        #pragma unroll
        for (int i = 0; i < 3; ++i) {
            f4 x = *(const f4*)(srow0 + (size_t)r * SEQ + (i * 32 + lane) * 4);
            x += bv[i];
            sv[i] = x;
            m = fmaxf(m, fmaxf(fmaxf(x[0], x[1]), fmaxf(x[2], x[3])));
        }
        #pragma unroll
        for (int off = 16; off > 0; off >>= 1)
            m = fmaxf(m, __shfl_xor(m, off, 32));

        float sum = 0.f;
        #pragma unroll
        for (int i = 0; i < 3; ++i) {
            float e0 = exp2f((sv[i][0] - m) * LOG2E);
            float e1 = exp2f((sv[i][1] - m) * LOG2E);
            float e2 = exp2f((sv[i][2] - m) * LOG2E);
            float e3 = exp2f((sv[i][3] - m) * LOG2E);
            sum += (e0 + e1) + (e2 + e3);
            h4 eh = { (_Float16)e0, (_Float16)e1, (_Float16)e2, (_Float16)e3 };
            *(h4*)(prow_base + r * P_STRIDE + (i * 32 + lane) * 4) = eh;
        }
        #pragma unroll
        for (int off = 16; off > 0; off >>= 1)
            sum += __shfl_xor(sum, off, 32);
        if (lane == 0) s_rowinv[w * 16 + r] = 1.0f / sum;
    }
    // probs/rowinv are wave-private; same-wave LDS RAW is handled by compiler DScnt waits.

    // ---- Stage 3: probs(16xS f16) @ Vt -> 16x64 f32 via v_wmma_f32_16x16x32_f16
    const int hlf = lane >> 4;     // half-wave select
    const int n   = lane & 15;     // A row (M) and B column (N)

    v8f acc[4] = {};               // 4 D-tiles of 16 columns
    const _Float16* prow = prow_base + n * P_STRIDE;

    #pragma unroll 2
    for (int c = 0; c < 12; ++c) {                 // K chunks of 32
        const int kA = c * 32 + hlf * 8;
        h8 alo = *(const h8*)(prow + kA);          // K = kA .. kA+7
        h8 ahi = *(const h8*)(prow + kA + 16);     // K = kA+16 .. kA+23
        v16h a = __builtin_shufflevector(alo, ahi,
            0,1,2,3,4,5,6,7,8,9,10,11,12,13,14,15);

        const int kB = c * 32 + hlf * 8;
        #pragma unroll
        for (int dt = 0; dt < 4; ++dt) {
            const _Float16* vrow = &s_vt[(dt * 16 + n) * VT_STRIDE];
            h8 blo = *(const h8*)(vrow + kB);      // K = kB .. kB+7
            h8 bhi = *(const h8*)(vrow + kB + 16); // K = kB+16 .. kB+23
            v16h b = __builtin_shufflevector(blo, bhi,
                0,1,2,3,4,5,6,7,8,9,10,11,12,13,14,15);
            acc[dt] = __builtin_amdgcn_wmma_f32_16x16x32_f16(
                false, a, false, b, (short)0, acc[dt], false, false);
        }
    }

    // ---- Stage 4: normalize by rowsum and store fp32 output
    float inv[8];
    #pragma unroll
    for (int v = 0; v < 8; ++v)
        inv[v] = s_rowinv[w * 16 + hlf * 8 + v];

    float* obase = out + ((size_t)bh * SEQ + q0 + w * 16) * DIM;
    #pragma unroll
    for (int dt = 0; dt < 4; ++dt) {
        #pragma unroll
        for (int v = 0; v < 8; ++v) {
            // C/D layout: lanes 0-15 -> M=v, lanes 16-31 -> M=v+8; N = lane&15
            obase[(size_t)(hlf * 8 + v) * DIM + dt * 16 + n] = acc[dt][v] * inv[v];
        }
    }
}

extern "C" void kernel_launch(void* const* d_in, const int* in_sizes, int n_in,
                              void* d_out, int out_size, void* d_ws, size_t ws_size,
                              hipStream_t stream) {
    const float* scores = (const float*)d_in[0];  // [64,4,384,384]
    const float* V      = (const float*)d_in[1];  // [64,4,384,64]
    const float* bias   = (const float*)d_in[2];  // [384]
    float* out          = (float*)d_out;          // [64,4,384,64]

    const int BH = 64 * 4;
    dim3 grid(BH * (SEQ / 64));   // 1536 blocks
    dim3 block(128);              // 4 waves; each wave = one 16x64 output tile
    attn_pv_kernel<<<grid, block, 0, stream>>>(scores, V, bias, out);
}
